// LocalSelfAttention_18554258719261
// MI455X (gfx1250) — matrix operand order repaired and checked by reading, not compiled, not run on hardware
//
#include <hip/hip_runtime.h>
#include <hip/hip_bf16.h>

// ---------------- problem constants ----------------
#define S_  2048
#define B_  4
#define D_  1024
#define H_  16
#define DH_ 64
#define W2_ 128
#define BS_ (B_*S_)
#define NEGF (-3.402823466e38f)

typedef _Float16 half16v __attribute__((ext_vector_type(16)));
typedef _Float16 half8v  __attribute__((ext_vector_type(8)));
typedef float    f32x8   __attribute__((ext_vector_type(8)));
typedef unsigned int uint32x4v __attribute__((ext_vector_type(4)));
typedef int      int32x4v __attribute__((ext_vector_type(4)));
typedef int      int32x8v __attribute__((ext_vector_type(8)));

#define HAVE_TDM __has_builtin(__builtin_amdgcn_tensor_load_to_lds)

// ---------------------------------------------------------------------------
// h = x.transpose(1,0,2) converted to f16 row-major [B*S, D]  (WMMA A operand)
// ---------------------------------------------------------------------------
__global__ void hconv_kernel(const float* __restrict__ x, _Float16* __restrict__ hf) {
    int i  = blockIdx.x * 256 + threadIdx.x;      // (b*S+s)*D + k
    int kk = i % D_;
    int m  = i / D_;
    int s  = m % S_;
    int b  = m / S_;
    hf[i] = (_Float16)x[((size_t)s * B_ + b) * D_ + kk];
}

// ---------------------------------------------------------------------------
// Pack W [N,K] f32 -> WMMA B-fragment-major f16.
// Fragment (kb, nt): lane l, elem e holds W[nt*16+(l&15)][kb*32+16*(l>>4)+e]
// ---------------------------------------------------------------------------
__global__ void packw_kernel(const float* __restrict__ Wsrc, _Float16* __restrict__ dst) {
    int i  = blockIdx.x * 256 + threadIdx.x;      // D*D elements
    int e  = i & 15;
    int l  = (i >> 4) & 31;
    int nt = (i >> 9) & 63;
    int kb = i >> 15;
    int n  = nt * 16 + (l & 15);
    int kk = kb * 32 + 16 * (l >> 4) + e;
    dst[i] = (_Float16)Wsrc[(size_t)n * D_ + kk];
}

// ---------------------------------------------------------------------------
// WMMA GEMM: Y[M=8192, N=1024] = A_f16[M,K=1024] * Bpack + bias, *scale.
// Block = 256 threads (8 waves) computes a 128M x 64N tile.
// B fragments for the 64-col supertile are staged into LDS in two 64 KB
// phases by the Tensor Data Mover (TDM); all 8 waves consume them from LDS.
// ---------------------------------------------------------------------------
__global__ __launch_bounds__(256) void gemm16_kernel(
    const _Float16* __restrict__ A, const _Float16* __restrict__ Bp,
    const float* __restrict__ bias, float scale,
    _Float16* __restrict__ outH, float* __restrict__ outF)
{
    __shared__ __align__(64) _Float16 bsh[16 * 2048];   // 64 KB: 16 K-blocks x 4 frag-tiles

    const int lane   = threadIdx.x & 31;
    const int wv     = threadIdx.x >> 5;
    const int n4     = blockIdx.x & 15;                 // 64-col supertile
    const int mtile  = (blockIdx.x >> 4) * 8 + wv;      // 512 m-tiles total
    const int laneLo = lane & 15, laneHi = lane >> 4;

    f32x8 zero = {};
    f32x8 acc[4] = {zero, zero, zero, zero};
    const _Float16* arow = A + (size_t)(mtile * 16 + laneLo) * D_;

    for (int ph = 0; ph < 2; ++ph) {
        // ---- stage B rows [ph*16, ph*16+16) for this supertile into LDS ----
#if HAVE_TDM
        if (threadIdx.x < 32) {
            unsigned long long ga = (unsigned long long)(uintptr_t)Bp +
                (((unsigned long long)ph * 16ull) * 32768ull +
                 (unsigned long long)n4 * 2048ull) * 2ull;       // byte address
            unsigned int lds = (unsigned int)(uintptr_t)(void*)bsh;
            uint32x4v g0;
            g0[0] = 1u;                                          // count=1, user mode
            g0[1] = lds;                                         // lds_addr
            g0[2] = (unsigned int)(ga & 0xffffffffull);          // global_addr lo
            g0[3] = (unsigned int)((ga >> 32) & 0x01fffffful) | (2u << 30); // hi | type=2
            int32x8v g1;
            g1[0] = 1 << 16;                      // data_size=1 (2 bytes/elem)
            g1[1] = (2048 & 0xffff) << 16;        // tensor_dim0[15:0] (bits 63:48)
            g1[2] = (16 & 0xffff) << 16;          // dim0 hi=0 | tensor_dim1 lo=16
            g1[3] = 2048 << 16;                   // tensor_dim1 hi=0 | tile_dim0=2048
            g1[4] = 16;                           // tile_dim1=16, tile_dim2=0
            g1[5] = 32768;                        // tensor_dim0_stride lo (elements)
            g1[6] = 0;                            // stride hi | dim1_stride lo
            g1[7] = 0;
            int32x4v gz = {};
#if __clang_major__ >= 23
            int32x8v gz8 = {};
            __builtin_amdgcn_tensor_load_to_lds(g0, g1, gz, gz, gz8, 0);
#else
            __builtin_amdgcn_tensor_load_to_lds(g0, g1, gz, gz, 0);
#endif
            __builtin_amdgcn_s_wait_tensorcnt(0);
        }
        __syncthreads();
#else
        {   // cooperative fallback copy: 16 rows x 4 KB, rows strided 64 KB
            const _Float16* src = Bp + ((size_t)ph * 16 * 32768 + (size_t)n4 * 2048);
            for (int c = threadIdx.x; c < 4096; c += 256) {      // half8 chunks
                int row  = c >> 8;                               // 256 chunks/row
                int col8 = c & 255;
                *(half8v*)&bsh[row * 2048 + col8 * 8] =
                    *(const half8v*)&src[(size_t)row * 32768 + col8 * 8];
            }
        }
        __syncthreads();
#endif
        // ---- 16 K-steps: A from global, B from LDS ----
        for (int kk = 0; kk < 16; ++kk) {
            int kb = ph * 16 + kk;
            const _Float16* ap = arow + kb * 32 + 16 * laneHi;
            __builtin_prefetch(arow + (kb + 1) * 32, 0, 0);
            half8v lo = *(const half8v*)(ap);
            half8v hi = *(const half8v*)(ap + 8);
            half16v a;
            #pragma unroll
            for (int i = 0; i < 8; ++i) { a[i] = lo[i]; a[i + 8] = hi[i]; }
            #pragma unroll
            for (int t = 0; t < 4; ++t) {
                half16v b = *(const half16v*)&bsh[kk * 2048 + (t * 32 + lane) * 16];
                acc[t] = __builtin_amdgcn_wmma_f32_16x16x32_f16(
                    false, a, false, b, (short)0, acc[t], false, false);
            }
        }
        __syncthreads();    // protect LDS before next phase overwrites it
    }

    #pragma unroll
    for (int t = 0; t < 4; ++t) {
        int n = (n4 * 4 + t) * 16 + laneLo;
        float bv = bias[n];
        #pragma unroll
        for (int r = 0; r < 8; ++r) {
            int mm = mtile * 16 + r + 8 * laneHi;
            float v = (acc[t][r] + bv) * scale;
            if (outF) outF[(size_t)mm * D_ + n] = v;
            if (outH) outH[(size_t)mm * D_ + n] = (_Float16)v;
        }
    }
}

// ---------------------------------------------------------------------------
// v [B,S,H,DH] f16 -> vT [B,H,DH,S] f16 (contiguous-key B operand for PV)
// ---------------------------------------------------------------------------
__global__ void vtrans_kernel(const _Float16* __restrict__ v, _Float16* __restrict__ vT) {
    int i  = blockIdx.x * 256 + threadIdx.x;   // vT linear index
    int s  = i % S_;
    int t  = i / S_;
    int dh = t % DH_;
    int t2 = t / DH_;
    int hh = t2 % H_;
    int b  = t2 / H_;
    vT[i] = v[(((size_t)b * S_ + s) * H_ + hh) * DH_ + dh];
}

// ---------------------------------------------------------------------------
// qg[b, n] = (x[0,b,:] . Wqg[n,:] + bqg[n]) * 1/sqrt(DH)
// ---------------------------------------------------------------------------
__global__ __launch_bounds__(256) void qg_kernel(
    const float* __restrict__ x, const float* __restrict__ Wqg,
    const float* __restrict__ bqg, float* __restrict__ qg)
{
    int i = blockIdx.x * 256 + threadIdx.x;    // b*D + n
    int b = i / D_, n = i % D_;
    const float* xr = x + (size_t)b * D_;      // x[s=0, b, :]
    const float* wr = Wqg + (size_t)n * D_;
    float acc = bqg[n];
    for (int kk = 0; kk < D_; ++kk) acc += xr[kk] * wr[kk];
    qg[i] = acc * 0.125f;
}

// ---------------------------------------------------------------------------
// Banded local attention + global column; one wave per (b, h, 16-query tile).
// QK^T and P*V via v_wmma_f32_16x16x32_f16; softmax in f32 through LDS.
// ---------------------------------------------------------------------------
__global__ __launch_bounds__(64) void attn_kernel(
    const _Float16* __restrict__ q, const _Float16* __restrict__ k,
    const _Float16* __restrict__ vT, const unsigned char* __restrict__ kpm,
    _Float16* __restrict__ attnH)
{
    __shared__ float    sc[2][16][304];    // 288 window cols + col 288 = global
    __shared__ _Float16 pl[2][16][288];    // probs (f16) for PV WMMA
    __shared__ float    pgs[2][16];        // global-column prob

    const int lane   = threadIdx.x & 31;
    const int w      = threadIdx.x >> 5;
    const int task   = blockIdx.x * 2 + w;
    const int b      = task / (H_ * (S_ / 16));
    const int rem    = task % (H_ * (S_ / 16));
    const int hh     = rem / (S_ / 16);
    const int q0     = (rem % (S_ / 16)) * 16;
    const int ks     = q0 - W2_;
    const int laneLo = lane & 15, laneHi = lane >> 4;

    // --- A fragments for this query tile (2 K-chunks over DH=64), reused ---
    half16v aq[2];
    {
        const _Float16* qp = q + (((size_t)b * S_ + q0 + laneLo) * H_ + hh) * DH_;
        #pragma unroll
        for (int c = 0; c < 2; ++c) {
            const _Float16* p = qp + c * 32 + 16 * laneHi;
            half8v lo = *(const half8v*)p;
            half8v hi = *(const half8v*)(p + 8);
            #pragma unroll
            for (int i = 0; i < 8; ++i) { aq[c][i] = lo[i]; aq[c][i + 8] = hi[i]; }
        }
    }

    // --- scores: 18 key-tiles of 16 covering [ks, ks+287] ---
    for (int kt = 0; kt < 18; ++kt) {
        int key0 = ks + kt * 16;
        int key  = key0 + laneLo;
        int keyc = min(max(key, 0), S_ - 1);
        const _Float16* kp = k + (((size_t)b * S_ + keyc) * H_ + hh) * DH_;
        f32x8 accs = {};
        #pragma unroll
        for (int c = 0; c < 2; ++c) {
            half16v bfr = *(const half16v*)(kp + c * 32 + 16 * laneHi);
            accs = __builtin_amdgcn_wmma_f32_16x16x32_f16(
                false, aq[c], false, bfr, (short)0, accs, false, false);
        }
        #pragma unroll
        for (int r = 0; r < 8; ++r) {
            int m     = r + 8 * laneHi;
            int query = q0 + m;
            int kpos  = key0 + laneLo;
            bool valid = (kpos >= 0) && (kpos < S_) &&
                         (kpos - query >= -W2_) && (kpos - query <= W2_);
            float sv;
            if (!valid) {
                sv = -__builtin_inff();
            } else {
                sv = accs[r];
                bool madd = (kpos == 0) || (kpm[b * S_ + kpos] && kpos != 0);
                if (madd) sv += NEGF;          // is_global | is_masked removed
            }
            sc[w][m][kt * 16 + laneLo] = sv;
        }
    }
    // --- global column: q . k[b,0,h,:] ---
    if (lane < 16) {
        const _Float16* qp = q + (((size_t)b * S_ + q0 + lane) * H_ + hh) * DH_;
        const _Float16* k0 = k + ((size_t)b * S_ * H_ + hh) * DH_;
        float g = 0.f;
        for (int d = 0; d < DH_; ++d) g += (float)qp[d] * (float)k0[d];
        sc[w][lane][288] = g;
    }
    __syncthreads();

    // --- softmax over 289 columns (lanes 0..15 own one row each) ---
    if (lane < 16) {
        int m = lane, query = q0 + m;
        bool qmask = (kpm[b * S_ + query] != 0) && (query != 0);
        float mx = -__builtin_inff();
        for (int j = 0; j <= 288; ++j) mx = fmaxf(mx, sc[w][m][j]);
        float sum = 0.f;
        for (int j = 0; j <= 288; ++j) sum += __expf(sc[w][m][j] - mx);
        float inv = 1.0f / sum;
        for (int j = 0; j < 288; ++j) {
            float p = qmask ? 0.f : __expf(sc[w][m][j] - mx) * inv;
            pl[w][m][j] = (_Float16)p;
        }
        pgs[w][m] = qmask ? 0.f : __expf(sc[w][m][288] - mx) * inv;
    }
    __syncthreads();

    // --- PV: 9 K-chunks of 32 keys x 4 DH-tiles ---
    f32x8 zero = {};
    f32x8 acc[4] = {zero, zero, zero, zero};
    for (int cc = 0; cc < 9; ++cc) {
        half16v ap;
        {
            const _Float16* pp = &pl[w][laneLo][cc * 32 + 16 * laneHi];
            half8v lo = *(const half8v*)pp;
            half8v hi = *(const half8v*)(pp + 8);
            #pragma unroll
            for (int i = 0; i < 8; ++i) { ap[i] = lo[i]; ap[i + 8] = hi[i]; }
        }
        int keybase = ks + cc * 32 + 16 * laneHi;
        keybase = min(max(keybase, 0), S_ - 16);   // OOB keys have p==0
        #pragma unroll
        for (int t = 0; t < 4; ++t) {
            int dh = t * 16 + laneLo;
            const _Float16* vp =
                vT + (((size_t)b * H_ + hh) * DH_ + dh) * S_ + keybase;
            half16v bfr = *(const half16v*)vp;
            acc[t] = __builtin_amdgcn_wmma_f32_16x16x32_f16(
                false, ap, false, bfr, (short)0, acc[t], false, false);
        }
    }
    // --- epilogue: add global-key contribution, store f16 [B,S,H,DH] ---
    #pragma unroll
    for (int t = 0; t < 4; ++t) {
        int dh = t * 16 + laneLo;
        float v0 = (float)vT[(((size_t)b * H_ + hh) * DH_ + dh) * S_];  // v[b,0,h,dh]
        #pragma unroll
        for (int r = 0; r < 8; ++r) {
            int m = r + 8 * laneHi;
            float val = acc[t][r] + pgs[w][m] * v0;
            attnH[(((size_t)b * S_ + q0 + m) * H_ + hh) * DH_ + dh] = (_Float16)val;
        }
    }
}

// ---------------------------------------------------------------------------
// Full attention for the global token (query position 0) with qg/kg/vg.
// One block per (b, h). Overwrites attn row s==0.
// ---------------------------------------------------------------------------
__global__ __launch_bounds__(256) void global_attn_kernel(
    const float* __restrict__ qg, const _Float16* __restrict__ kg,
    const _Float16* __restrict__ vg, const unsigned char* __restrict__ kpm,
    _Float16* __restrict__ attnH)
{
    __shared__ float scb[S_];
    __shared__ float red[256];
    __shared__ float qv[DH_];
    const int b   = blockIdx.x / H_;
    const int hh  = blockIdx.x % H_;
    const int tid = threadIdx.x;
    if (tid < DH_) qv[tid] = qg[b * D_ + hh * DH_ + tid];
    __syncthreads();

    for (int s = tid; s < S_; s += 256) {
        bool masked = (kpm[b * S_ + s] != 0) && (s != 0);
        float d = 0.f;
        const _Float16* kp = kg + (((size_t)b * S_ + s) * H_ + hh) * DH_;
        for (int j = 0; j < DH_; ++j) d += qv[j] * (float)kp[j];
        scb[s] = masked ? NEGF : d;
    }
    __syncthreads();
    float mx = -__builtin_inff();
    for (int s = tid; s < S_; s += 256) mx = fmaxf(mx, scb[s]);
    red[tid] = mx; __syncthreads();
    for (int o = 128; o > 0; o >>= 1) { if (tid < o) red[tid] = fmaxf(red[tid], red[tid + o]); __syncthreads(); }
    mx = red[0]; __syncthreads();
    float sum = 0.f;
    for (int s = tid; s < S_; s += 256) { float e = __expf(scb[s] - mx); scb[s] = e; sum += e; }
    __syncthreads();
    red[tid] = sum; __syncthreads();
    for (int o = 128; o > 0; o >>= 1) { if (tid < o) red[tid] += red[tid + o]; __syncthreads(); }
    float inv = 1.0f / red[0];
    __syncthreads();
    if (tid < DH_) {
        float accv = 0.f;
        const _Float16* vp = vg + ((size_t)b * S_ * H_ + hh) * DH_ + tid;
        for (int s = 0; s < S_; ++s) accv += scb[s] * (float)vp[(size_t)s * H_ * DH_];
        attnH[((size_t)b * S_ * H_ + hh) * DH_ + tid] = (_Float16)(accv * inv);
    }
}

// ---------------------------------------------------------------------------
// Residual + LayerNorm; writes output [S,B,D] f32. One block per (b,s) row.
// ---------------------------------------------------------------------------
__global__ __launch_bounds__(256) void ln_kernel(
    const float* __restrict__ y, const float* __restrict__ x,
    const float* __restrict__ g, const float* __restrict__ bb,
    float* __restrict__ out)
{
    __shared__ float red[256];
    const int row = blockIdx.x;            // b*S + s
    const int b = row / S_, s = row % S_;
    const int tid = threadIdx.x;
    float vals[4];
    float sum = 0.f;
    #pragma unroll
    for (int j = 0; j < 4; ++j) {
        int d = tid + j * 256;
        float v = y[(size_t)row * D_ + d] + x[((size_t)s * B_ + b) * D_ + d];
        vals[j] = v; sum += v;
    }
    red[tid] = sum; __syncthreads();
    for (int o = 128; o > 0; o >>= 1) { if (tid < o) red[tid] += red[tid + o]; __syncthreads(); }
    float mu = red[0] * (1.0f / D_); __syncthreads();
    float vs = 0.f;
    #pragma unroll
    for (int j = 0; j < 4; ++j) { float d0 = vals[j] - mu; vs += d0 * d0; }
    red[tid] = vs; __syncthreads();
    for (int o = 128; o > 0; o >>= 1) { if (tid < o) red[tid] += red[tid + o]; __syncthreads(); }
    float rstd = rsqrtf(red[0] * (1.0f / D_) + 1e-5f);
    #pragma unroll
    for (int j = 0; j < 4; ++j) {
        int d = tid + j * 256;
        out[((size_t)s * B_ + b) * D_ + d] = (vals[j] - mu) * rstd * g[d] + bb[d];
    }
}

// ---------------------------------------------------------------------------
extern "C" void kernel_launch(void* const* d_in, const int* in_sizes, int n_in,
                              void* d_out, int out_size, void* d_ws, size_t ws_size,
                              hipStream_t stream) {
    const float* x    = (const float*)d_in[0];
    const unsigned char* kpm = (const unsigned char*)d_in[1];
    const float* Wq  = (const float*)d_in[2];  const float* bq  = (const float*)d_in[3];
    const float* Wk  = (const float*)d_in[4];  const float* bk  = (const float*)d_in[5];
    const float* Wv  = (const float*)d_in[6];  const float* bv  = (const float*)d_in[7];
    const float* Wqg = (const float*)d_in[8];  const float* bqg = (const float*)d_in[9];
    const float* Wkg = (const float*)d_in[10]; const float* bkg = (const float*)d_in[11];
    const float* Wvg = (const float*)d_in[12]; const float* bvg = (const float*)d_in[13];
    const float* Wo  = (const float*)d_in[14]; const float* bo  = (const float*)d_in[15];
    const float* lng = (const float*)d_in[16]; const float* lnb = (const float*)d_in[17];
    float* out = (float*)d_out;

    // ---- workspace carve-up ----
    char* base = (char*)d_ws;
    size_t off = 0;
    auto alloc = [&](size_t bytes) { size_t r = off; off += (bytes + 255) & ~(size_t)255; return r; };
    const size_t ACT16 = (size_t)BS_ * D_ * sizeof(_Float16);  // 16 MB
    const size_t WPK   = (size_t)D_ * D_ * sizeof(_Float16);   //  2 MB

    _Float16* hf16   = (_Float16*)(base + alloc(ACT16));
    _Float16* pWq    = (_Float16*)(base + alloc(WPK));
    _Float16* pWk    = (_Float16*)(base + alloc(WPK));
    _Float16* pWv    = (_Float16*)(base + alloc(WPK));
    _Float16* pWkg   = (_Float16*)(base + alloc(WPK));
    _Float16* pWvg   = (_Float16*)(base + alloc(WPK));
    _Float16* pWo    = (_Float16*)(base + alloc(WPK));
    _Float16* qf16   = (_Float16*)(base + alloc(ACT16));
    _Float16* kf16   = (_Float16*)(base + alloc(ACT16));
    _Float16* vf16   = (_Float16*)(base + alloc(ACT16));
    _Float16* vT16   = (_Float16*)(base + alloc(ACT16));
    _Float16* kg16   = (_Float16*)(base + alloc(ACT16));
    _Float16* vg16   = (_Float16*)(base + alloc(ACT16));
    _Float16* at16   = (_Float16*)(base + alloc(ACT16));
    float*    qgf    = (float*)(base + alloc((size_t)B_ * D_ * sizeof(float)));
    float*    yf32   = (float*)(base + alloc((size_t)BS_ * D_ * sizeof(float)));
    (void)ws_size; (void)in_sizes; (void)n_in; (void)out_size;

    const float qscale = 0.125f;  // 1/sqrt(64)

    // 1. layout/precision conversions
    hipLaunchKernelGGL(hconv_kernel, dim3(BS_ * D_ / 256), dim3(256), 0, stream, x, hf16);
    hipLaunchKernelGGL(packw_kernel, dim3(D_ * D_ / 256), dim3(256), 0, stream, Wq,  pWq);
    hipLaunchKernelGGL(packw_kernel, dim3(D_ * D_ / 256), dim3(256), 0, stream, Wk,  pWk);
    hipLaunchKernelGGL(packw_kernel, dim3(D_ * D_ / 256), dim3(256), 0, stream, Wv,  pWv);
    hipLaunchKernelGGL(packw_kernel, dim3(D_ * D_ / 256), dim3(256), 0, stream, Wkg, pWkg);
    hipLaunchKernelGGL(packw_kernel, dim3(D_ * D_ / 256), dim3(256), 0, stream, Wvg, pWvg);
    hipLaunchKernelGGL(packw_kernel, dim3(D_ * D_ / 256), dim3(256), 0, stream, Wo,  pWo);

    // 2. projection GEMMs (WMMA + TDM-staged B): 1024 blocks x 8 waves
    const dim3 ggrid(1024), gblk(256);
    hipLaunchKernelGGL(gemm16_kernel, ggrid, gblk, 0, stream, hf16, pWq,  bq,  qscale, qf16, (float*)nullptr);
    hipLaunchKernelGGL(gemm16_kernel, ggrid, gblk, 0, stream, hf16, pWk,  bk,  1.0f,   kf16, (float*)nullptr);
    hipLaunchKernelGGL(gemm16_kernel, ggrid, gblk, 0, stream, hf16, pWv,  bv,  1.0f,   vf16, (float*)nullptr);
    hipLaunchKernelGGL(gemm16_kernel, ggrid, gblk, 0, stream, hf16, pWkg, bkg, 1.0f,   kg16, (float*)nullptr);
    hipLaunchKernelGGL(gemm16_kernel, ggrid, gblk, 0, stream, hf16, pWvg, bvg, 1.0f,   vg16, (float*)nullptr);

    // 3. v transpose for PV B-operand; qg matvec
    hipLaunchKernelGGL(vtrans_kernel, dim3(BS_ * D_ / 256), dim3(256), 0, stream, vf16, vT16);
    hipLaunchKernelGGL(qg_kernel, dim3(B_ * D_ / 256), dim3(256), 0, stream, x, Wqg, bqg, qgf);

    // 4. banded attention (WMMA QK^T + PV), then global-token attention
    hipLaunchKernelGGL(attn_kernel, dim3(B_ * H_ * (S_ / 16) / 2), dim3(64), 0, stream,
                       qf16, kf16, vT16, kpm, at16);
    hipLaunchKernelGGL(global_attn_kernel, dim3(B_ * H_), dim3(256), 0, stream,
                       qgf, kg16, vg16, kpm, at16);

    // 5. output projection (WMMA) + residual LayerNorm
    hipLaunchKernelGGL(gemm16_kernel, ggrid, gblk, 0, stream, at16, pWo, bo, 1.0f,
                       (_Float16*)nullptr, yf32);
    hipLaunchKernelGGL(ln_kernel, dim3(BS_), dim3(256), 0, stream, yf32, x, lng, lnb, out);
}
